// ExpNegL2_74655121539841
// MI455X (gfx1250) — matrix operand order repaired and checked by reading, not compiled, not run on hardware
//
#include <hip/hip_runtime.h>
#include <hip/hip_bf16.h>
#include <math.h>
#include <stdint.h>

typedef _Float16 v8h  __attribute__((ext_vector_type(8)));
typedef _Float16 v16h __attribute__((ext_vector_type(16)));
typedef float    v8f  __attribute__((ext_vector_type(8)));

#define D_K    512     // feature dim (= K of GEMM)
#define M_ROWS 8192    // 64 * 128 audio rows
#define N_ROWS 2048    // 64 * 32 text rows
#define BM     128
#define BN     256
#define BK     32

// ---------------------------------------------------------------------------
// Async global->LDS (GLOBAL_LOAD_ASYNC_TO_LDS_B128, ASYNCcnt) with sync fallback.
// Builtin prototype (from hipcc diagnostic): param 1 is
//   int __attribute__((vector_size(16))) addrspace(1)*  (printed as "__device__ *")
// so pass int4-vector pointers with explicit address spaces.
// ---------------------------------------------------------------------------
#if __has_builtin(__builtin_amdgcn_global_load_async_to_lds_b128)
#define HAVE_ASYNC_LDS 1
typedef int v4i_vs __attribute__((vector_size(16)));
typedef __attribute__((address_space(1))) v4i_vs glob_v4i;
typedef __attribute__((address_space(3))) v4i_vs lds_v4i;
__device__ __forceinline__ void cp16_g2l(const _Float16* g, _Float16* l) {
    __builtin_amdgcn_global_load_async_to_lds_b128(
        (glob_v4i*)(uintptr_t)g,
        (lds_v4i*)(unsigned int)(uintptr_t)l,
        0, 0);
}
__device__ __forceinline__ void wait_async_done() {
#if __has_builtin(__builtin_amdgcn_s_wait_asynccnt)
    __builtin_amdgcn_s_wait_asynccnt(0);
#else
    asm volatile("s_wait_asynccnt 0x0" ::: "memory");
#endif
}
#else
#define HAVE_ASYNC_LDS 0
__device__ __forceinline__ void cp16_g2l(const _Float16* g, _Float16* l) {
    *(v8h*)l = *(const v8h*)g;
}
__device__ __forceinline__ void wait_async_done() {}
#endif

// ---------------------------------------------------------------------------
// Kernel 1: L2-normalize rows of f32 [rows x 512] and convert to f16.
// One wave per row; wave32 shuffle reduction for the squared norm.
// ---------------------------------------------------------------------------
__global__ __launch_bounds__(256)
void normalize_rows_f16(const float* __restrict__ src,
                        _Float16* __restrict__ dst, int rows) {
    const int wave = threadIdx.x >> 5;
    const int lane = threadIdx.x & 31;
    const int row  = blockIdx.x * 8 + wave;
    if (row >= rows) return;

    const float* p = src + (size_t)row * D_K + lane * 16;
    float4 x0 = ((const float4*)p)[0];
    float4 x1 = ((const float4*)p)[1];
    float4 x2 = ((const float4*)p)[2];
    float4 x3 = ((const float4*)p)[3];

    float ss = x0.x*x0.x + x0.y*x0.y + x0.z*x0.z + x0.w*x0.w
             + x1.x*x1.x + x1.y*x1.y + x1.z*x1.z + x1.w*x1.w
             + x2.x*x2.x + x2.y*x2.y + x2.z*x2.z + x2.w*x2.w
             + x3.x*x3.x + x3.y*x3.y + x3.z*x3.z + x3.w*x3.w;

    #pragma unroll
    for (int off = 16; off > 0; off >>= 1)
        ss += __shfl_xor(ss, off, 32);

    const float scale = 1.0f / fmaxf(sqrtf(ss), 1e-12f);

    v8h h0, h1;
    h0[0] = (_Float16)(x0.x * scale); h0[1] = (_Float16)(x0.y * scale);
    h0[2] = (_Float16)(x0.z * scale); h0[3] = (_Float16)(x0.w * scale);
    h0[4] = (_Float16)(x1.x * scale); h0[5] = (_Float16)(x1.y * scale);
    h0[6] = (_Float16)(x1.z * scale); h0[7] = (_Float16)(x1.w * scale);
    h1[0] = (_Float16)(x2.x * scale); h1[1] = (_Float16)(x2.y * scale);
    h1[2] = (_Float16)(x2.z * scale); h1[3] = (_Float16)(x2.w * scale);
    h1[4] = (_Float16)(x3.x * scale); h1[5] = (_Float16)(x3.y * scale);
    h1[6] = (_Float16)(x3.z * scale); h1[7] = (_Float16)(x3.w * scale);

    _Float16* q = dst + (size_t)row * D_K + lane * 16;
    ((v8h*)q)[0] = h0;
    ((v8h*)q)[1] = h1;
}

// ---------------------------------------------------------------------------
// Kernel 2: C = A (8192x512 f16) * B^T (2048x512 f16) via v_wmma_f32_16x16x32_f16.
// WG 256 threads (8 waves, 2x4). WG tile 128x256, wave tile 64x64 (4x4 WMMA).
// Double-buffered LDS (48 KB) with async global->LDS prefetch of tile k+1
// overlapped with the 16 WMMAs of tile k; one barrier per K-step.
// Epilogue: exp(-sqrt(max(2-2c,0))) scattered into the (b,i,s,j) layout.
// ---------------------------------------------------------------------------
__global__ __launch_bounds__(256)
void wmma_gemm_expneg(const _Float16* __restrict__ A,
                      const _Float16* __restrict__ B,
                      float* __restrict__ out) {
    __shared__ __align__(16) _Float16 sA[2][BM * BK];   // 2 x 8 KB
    __shared__ __align__(16) _Float16 sB[2][BN * BK];   // 2 x 16 KB

    const int tid  = threadIdx.x;
    const int wave = tid >> 5;
    const int lane = tid & 31;
    const int r    = lane & 15;   // row/col within 16-wide fragment
    const int h    = lane >> 4;   // half-wave selector

    const int m0 = blockIdx.x * BM;          // audio row base (one b-block)
    const int n0 = blockIdx.y * BN;          // text row base (8 i-blocks)
    const int waveM = (wave & 1) * 64;
    const int waveN = (wave >> 1) * 64;

    v8f acc[4][4];
    #pragma unroll
    for (int bm = 0; bm < 4; ++bm)
        #pragma unroll
        for (int bn = 0; bn < 4; ++bn)
            acc[bm][bn] = (v8f){0.f,0.f,0.f,0.f,0.f,0.f,0.f,0.f};

    // Stage one K-tile into LDS buffer `buf` (A: 2 x b128/thread, B: 4 x b128/thread).
    auto stage = [&](int buf, int kb) {
        #pragma unroll
        for (int t = 0; t < 2; ++t) {
            const int idx = tid + t * 256;        // 0..511
            const int row = idx >> 2;             // 0..127
            const int kc  = (idx & 3) * 8;        // 0,8,16,24
            cp16_g2l(&A[(size_t)(m0 + row) * D_K + kb + kc],
                     &sA[buf][row * BK + kc]);
        }
        #pragma unroll
        for (int t = 0; t < 4; ++t) {
            const int idx = tid + t * 256;        // 0..1023
            const int row = idx >> 2;             // 0..255
            const int kc  = (idx & 3) * 8;
            cp16_g2l(&B[(size_t)(n0 + row) * D_K + kb + kc],
                     &sB[buf][row * BK + kc]);
        }
    };

    stage(0, 0);
    wait_async_done();
    __syncthreads();

    for (int kb = 0; kb < D_K; kb += BK) {
        const int cur = (kb >> 5) & 1;
        const int nxt = cur ^ 1;
        if (kb + BK < D_K) stage(nxt, kb + BK);   // async prefetch next tile

        const _Float16* __restrict__ a_s = sA[cur];
        const _Float16* __restrict__ b_s = sB[cur];

        // A fragments (ISA 7.12.2, 16-bit A 16x32): lane r = row M;
        // VGPR0-3 <- K=h*8..+7, VGPR4-7 <- K=16+h*8..+7.
        v16h aF[4];
        #pragma unroll
        for (int bm = 0; bm < 4; ++bm) {
            const _Float16* base = &a_s[(waveM + bm * 16 + r) * BK];
            v8h lo = *(const v8h*)(base + h * 8);
            v8h hi = *(const v8h*)(base + 16 + h * 8);
            #pragma unroll
            for (int e = 0; e < 8; ++e) { aF[bm][e] = lo[e]; aF[bm][8 + e] = hi[e]; }
        }
        // B fragments: lane r = column N; same K order as A:
        // VGPR0-3 <- K=h*8..+7, VGPR4-7 <- K=16+h*8..+7.
        v16h bF[4];
        #pragma unroll
        for (int bn = 0; bn < 4; ++bn) {
            const _Float16* base = &b_s[(waveN + bn * 16 + r) * BK];
            v8h lo = *(const v8h*)(base + h * 8);
            v8h hi = *(const v8h*)(base + 16 + h * 8);
            #pragma unroll
            for (int e = 0; e < 8; ++e) { bF[bn][e] = lo[e]; bF[bn][8 + e] = hi[e]; }
        }

        #pragma unroll
        for (int bm = 0; bm < 4; ++bm)
            #pragma unroll
            for (int bn = 0; bn < 4; ++bn)
                acc[bm][bn] = __builtin_amdgcn_wmma_f32_16x16x32_f16(
                    false, aF[bm], false, bF[bn],
                    (short)0, acc[bm][bn], false, false);

        wait_async_done();   // my prefetch into nxt has landed in LDS
        __syncthreads();     // all waves done reading cur / staging nxt
    }

    // Epilogue. C/D layout: VGPR e, lanes 0-15 -> M=e, lanes 16-31 -> M=e+8;
    // N = lane%16. out[b,i,s,j] at b*262144 + i*4096 + s*32 + j.
    const int b = blockIdx.x;                     // M tile == one b block
    #pragma unroll
    for (int bm = 0; bm < 4; ++bm) {
        const int mmBase = waveM + bm * 16 + 8 * h;
        #pragma unroll
        for (int bn = 0; bn < 4; ++bn) {
            const int n_local = waveN + bn * 16 + r;          // 0..255
            const int i = blockIdx.y * 8 + (n_local >> 5);
            const int j = n_local & 31;
            float* o = out + (size_t)b * 262144 + (size_t)i * 4096 + j;
            #pragma unroll
            for (int e = 0; e < 8; ++e) {
                const int mm = mmBase + e;                    // s in 0..127
                const float v  = acc[bm][bn][e];
                const float d2 = fmaxf(2.0f - 2.0f * v, 0.0f);
                o[(size_t)mm * 32] = __expf(-__builtin_amdgcn_sqrtf(d2));
            }
        }
    }
}

// ---------------------------------------------------------------------------
extern "C" void kernel_launch(void* const* d_in, const int* in_sizes, int n_in,
                              void* d_out, int out_size, void* d_ws, size_t ws_size,
                              hipStream_t stream) {
    const float* audio = (const float*)d_in[0];   // (64,128,512) f32
    const float* text  = (const float*)d_in[1];   // (64,32,512)  f32
    float* out = (float*)d_out;                   // (64,64,128,32) f32

    _Float16* Af = (_Float16*)d_ws;               // 8192x512 f16 = 8 MB
    _Float16* Bf = Af + (size_t)M_ROWS * D_K;     // 2048x512 f16 = 2 MB

    normalize_rows_f16<<<M_ROWS / 8, 256, 0, stream>>>(audio, Af, M_ROWS);
    normalize_rows_f16<<<N_ROWS / 8, 256, 0, stream>>>(text,  Bf, N_ROWS);

    dim3 grid(M_ROWS / BM, N_ROWS / BN);          // 64 x 8 workgroups
    wmma_gemm_expneg<<<grid, dim3(256), 0, stream>>>(Af, Bf, out);
}
